// NaiveInductiveActorNetwork_14886356648085
// MI455X (gfx1250) — matrix operand order repaired and checked by reading, compiled, not run on hardware
//
#include <hip/hip_runtime.h>
#include <hip/hip_bf16.h>

typedef __bf16 bf16;
typedef bf16  bf16x8  __attribute__((ext_vector_type(8)));
typedef bf16  bf16x16 __attribute__((ext_vector_type(16)));
typedef float v8f     __attribute__((ext_vector_type(8)));

#define N_NODES   32768
#define F_IN      128
#define H1        256
#define H2        64
#define N_HOSTS   16384
#define N_GRAPHS  64
#define OUT_W     2050

#define USE_ASYNC_STAGE 1   // gfx1250 GLOBAL_LOAD_ASYNC_TO_LDS_B128 + ASYNCcnt

// ---------------------------------------------------------------- utilities
__global__ void k_cast_bf16(const float* __restrict__ s, bf16* __restrict__ d, int n) {
    int i = blockIdx.x * 256 + threadIdx.x;
    if (i < n) d[i] = (bf16)s[i];
}

// W [K x Nsrc] f32 row-major  ->  Wt [Npad x K] bf16 (transposed, zero-padded cols)
__global__ void k_transpose_cast(const float* __restrict__ W, bf16* __restrict__ Wt,
                                 int K, int Nsrc, int Npad) {
    int i = blockIdx.x * 256 + threadIdx.x;
    if (i >= Npad * K) return;
    int n = i / K, k = i - n * K;
    Wt[i] = (n < Nsrc) ? (bf16)W[(size_t)k * Nsrc + n] : (bf16)0.0f;
}

__global__ void k_pad_bias(const float* __restrict__ s, float* __restrict__ d,
                           int nsrc, int npad) {
    int i = blockIdx.x * 256 + threadIdx.x;
    if (i < npad) d[i] = (i < nsrc) ? s[i] : 0.0f;
}

__global__ void k_fill(float* __restrict__ p, float v, int n) {
    int i = blockIdx.x * 256 + threadIdx.x;
    if (i < n) p[i] = v;
}

__global__ void k_deg_add(const int* __restrict__ dst, int E, float* __restrict__ deg) {
    int e = blockIdx.x * 256 + threadIdx.x;
    if (e < E) unsafeAtomicAdd(&deg[dst[e]], 1.0f);
}

__global__ void k_rsqrt_inplace(float* __restrict__ p, int n) {
    int i = blockIdx.x * 256 + threadIdx.x;
    if (i < n) p[i] = rsqrtf(p[i]);
}

// ---- CDNA5 async global->LDS copy (16B per lane, tracked by ASYNCcnt) ----
__device__ __forceinline__ void async_copy_b128(void* lds_ptr, const void* gptr) {
    // LDS aperture: generic addr low 32 bits == LDS byte offset (ISA 10.2)
    unsigned int lds_off = (unsigned int)(unsigned long long)lds_ptr;
    asm volatile("global_load_async_to_lds_b128 %0, %1, off"
                 :: "v"(lds_off), "v"((unsigned long long)gptr)
                 : "memory");
}
__device__ __forceinline__ void wait_async0() {
    asm volatile("s_wait_asynccnt 0" ::: "memory");
}

// ------------------------------------------------- bf16 WMMA GEMM
// C = A[MxK](bf16,row-major) * Bt[NxK](bf16, pre-transposed N-major)
// K%32==0, M%64==0, N%(FN*32)==0.  Block: 128 threads = 4 waves (2x2).
// Wave tile: 32 x (FN*16).  Block tile: 64 x (FN*32).  Double-buffered LDS,
// next K-slice staged via global_load_async_to_lds_b128 while WMMAs run.
// Compile-time epilogue: row-scale, bias, relu, f32 write x2, bf16 write.
template <int FN, bool HAS_SCALE, bool HAS_BIAS, bool RELU,
          bool W_F, bool W_F2, bool W_B>
__global__ __launch_bounds__(128)
void k_gemm_bf16(const bf16* __restrict__ A, const bf16* __restrict__ Bt,
                 int M, int N, int K,
                 const float* __restrict__ rowscale, const float* __restrict__ bias,
                 float* __restrict__ outF, float* __restrict__ outF2,
                 bf16* __restrict__ outB) {
    constexpr int BN = FN * 32;                 // block tile N
    __shared__ bf16 Alds[2][64 * 32];
    __shared__ bf16 Blds[2][BN * 32];

    const int t    = threadIdx.x;
    const int lane = t & 31;
    const int wave = t >> 5;
    const int wrow = wave >> 1;                 // 0..1  (32 rows each)
    const int wcol = wave & 1;                  // 0..1  (FN*16 cols each)
    const int half = lane >> 4;                 // 0..1
    const int l16  = lane & 15;
    const int bm   = blockIdx.y, bn = blockIdx.x;

    v8f acc[2][FN] = {};

    // stage one 32-wide K slice of both tiles; fixed trip counts (no divergence)
    auto stage = [&](int buf, int k0) {
#if USE_ASYNC_STAGE
#pragma unroll
        for (int it = 0; it < 2; ++it) {         // A tile: 64 rows x 4 chunks
            int c = t + it * 128;
            int r = c >> 2, q = (c & 3) * 8;
            async_copy_b128(&Alds[buf][r * 32 + q],
                            A + (size_t)(bm * 64 + r) * K + k0 + q);
        }
#pragma unroll
        for (int it = 0; it < FN; ++it) {        // B tile: BN rows x 4 chunks
            int c = t + it * 128;
            int r = c >> 2, q = (c & 3) * 8;
            async_copy_b128(&Blds[buf][r * 32 + q],
                            Bt + (size_t)(bn * BN + r) * K + k0 + q);
        }
#else
        bf16x8 ta[2], tb[FN];
#pragma unroll
        for (int it = 0; it < 2; ++it) {
            int c = t + it * 128;
            int r = c >> 2, q = (c & 3) * 8;
            ta[it] = *(const bf16x8*)(A + (size_t)(bm * 64 + r) * K + k0 + q);
        }
#pragma unroll
        for (int it = 0; it < FN; ++it) {
            int c = t + it * 128;
            int r = c >> 2, q = (c & 3) * 8;
            tb[it] = *(const bf16x8*)(Bt + (size_t)(bn * BN + r) * K + k0 + q);
        }
#pragma unroll
        for (int it = 0; it < 2; ++it) {
            int c = t + it * 128;
            int r = c >> 2, q = (c & 3) * 8;
            *(bf16x8*)&Alds[buf][r * 32 + q] = ta[it];
        }
#pragma unroll
        for (int it = 0; it < FN; ++it) {
            int c = t + it * 128;
            int r = c >> 2, q = (c & 3) * 8;
            *(bf16x8*)&Blds[buf][r * 32 + q] = tb[it];
        }
#endif
    };

    auto compute = [&](int buf) {
        bf16x16 af[2], bfr[FN];
#pragma unroll
        for (int fm = 0; fm < 2; ++fm) {
            // A 16x32 frag: half 0 -> K 0..7 & 16..23, half 1 -> K 8..15 & 24..31
            int row = wrow * 32 + fm * 16 + l16;
            bf16x8 lo = *(const bf16x8*)&Alds[buf][row * 32 + half * 8];
            bf16x8 hi = *(const bf16x8*)&Alds[buf][row * 32 + half * 8 + 16];
#pragma unroll
            for (int i = 0; i < 8; ++i) { af[fm][i] = lo[i]; af[fm][i + 8] = hi[i]; }
        }
#pragma unroll
        for (int fn = 0; fn < FN; ++fn) {
            // B 32x16 frag: half 0 -> K 0..7 & 16..23, half 1 -> K 8..15 & 24..31 (N-major LDS)
            int col = wcol * (FN * 16) + fn * 16 + l16;
            bf16x8 lo = *(const bf16x8*)&Blds[buf][col * 32 + half * 8];
            bf16x8 hi = *(const bf16x8*)&Blds[buf][col * 32 + half * 8 + 16];
#pragma unroll
            for (int i = 0; i < 8; ++i) { bfr[fn][i] = lo[i]; bfr[fn][i + 8] = hi[i]; }
        }
#pragma unroll
        for (int fm = 0; fm < 2; ++fm)
#pragma unroll
            for (int fn = 0; fn < FN; ++fn)
                acc[fm][fn] = __builtin_amdgcn_wmma_f32_16x16x32_bf16(
                    false, af[fm], false, bfr[fn], (short)0, acc[fm][fn], false, false);
    };

    const int nsteps = K >> 5;
    stage(0, 0);
    for (int i = 0; i < nsteps; ++i) {
#if USE_ASYNC_STAGE
        wait_async0();                 // this wave's LDS writes landed
#endif
        __syncthreads();               // all waves' writes visible; prev reads retired
        if (i + 1 < nsteps) stage((i + 1) & 1, (i + 1) << 5);  // async prefetch
        compute(i & 1);                // WMMAs overlap the in-flight DMA
    }

    // ------------------------------ straight-line epilogue (no runtime branches)
    float rsv[2][8];
    if (HAS_SCALE) {
#pragma unroll
        for (int fm = 0; fm < 2; ++fm)
#pragma unroll
            for (int v = 0; v < 8; ++v)
                rsv[fm][v] = rowscale[bm * 64 + wrow * 32 + fm * 16 + half * 8 + v];
    }
    float bv[FN];
    if (HAS_BIAS) {
#pragma unroll
        for (int fn = 0; fn < FN; ++fn)
            bv[fn] = bias[bn * BN + wcol * (FN * 16) + fn * 16 + l16];
    }
#pragma unroll
    for (int fm = 0; fm < 2; ++fm)
#pragma unroll
        for (int fn = 0; fn < FN; ++fn)
#pragma unroll
            for (int v = 0; v < 8; ++v) {
                int row = bm * 64 + wrow * 32 + fm * 16 + half * 8 + v;
                int col = bn * BN + wcol * (FN * 16) + fn * 16 + l16;
                float val = acc[fm][fn][v];
                if (HAS_SCALE) val *= rsv[fm][v];
                if (HAS_BIAS)  val += bv[fn];
                if (RELU)      val = fmaxf(val, 0.0f);
                size_t idx = (size_t)row * N + col;
                if (W_F)  outF[idx]  = val;
                if (W_F2) outF2[idx] = val;
                if (W_B)  outB[idx]  = (bf16)val;
            }
}

// --------------------------------------------- edge scatter: agg[dst] += g[src]
// 16 floats per thread: 4x b128 gather + 16 global_atomic_add_f32.
__global__ void k_edge_scatter(const int* __restrict__ src, const int* __restrict__ dst,
                               const float* __restrict__ g, float* __restrict__ agg, int F) {
    int e = blockIdx.x * 256 + threadIdx.x;
    int f = blockIdx.y * 16;
    int s = src[e], d = dst[e];
    const float4* gp = (const float4*)(g + (size_t)s * F + f);
    float*        ap = agg + (size_t)d * F + f;
#pragma unroll
    for (int c = 0; c < 4; ++c) {
        float4 v = gp[c];
        unsafeAtomicAdd(ap + c * 4 + 0, v.x);
        unsafeAtomicAdd(ap + c * 4 + 1, v.y);
        unsafeAtomicAdd(ap + c * 4 + 2, v.z);
        unsafeAtomicAdd(ap + c * 4 + 3, v.w);
    }
}

// layer-1 post: hb1 = bf16(relu(agg*dinv[node] + b1)),  F == 256
__global__ void k_post1(const float* __restrict__ agg, const float* __restrict__ dinv,
                        const float* __restrict__ b, bf16* __restrict__ out) {
    int node = blockIdx.x, f = threadIdx.x;
    size_t i = (size_t)node * H1 + f;
    float v = agg[i] * dinv[node] + b[f];
    out[i] = (bf16)fmaxf(v, 0.0f);
}

// layer-2 post + host gather: hosts are nodes (i%512)<256, in order
__global__ void k_extract_hosts(const float* __restrict__ agg, const float* __restrict__ dinv,
                                const float* __restrict__ b, bf16* __restrict__ z) {
    int i = blockIdx.x * 256 + threadIdx.x;       // i < N_HOSTS * H2
    int r = i >> 6, f = i & 63;
    int node = (r >> 8) * 512 + (r & 255);
    float v = agg[(size_t)node * H2 + f] * dinv[node] + b[f];
    z[i] = (bf16)fmaxf(v, 0.0f);
}

// per-graph softmax over 256 nodes x 8 actions (transposed), + 2 leading zeros
__global__ __launch_bounds__(256)
void k_softmax(const float* __restrict__ act /* [N_HOSTS x 64], 8 valid cols */,
               float* __restrict__ out) {
    __shared__ float red[256];
    int g = blockIdx.x, t = threadIdx.x;
    const float* row = act + ((size_t)g * 256 + t) * 64;
    float v[8], m = -3.4e38f;
#pragma unroll
    for (int a = 0; a < 8; ++a) { v[a] = row[a]; m = fmaxf(m, v[a]); }
    red[t] = m; __syncthreads();
    for (int s = 128; s > 0; s >>= 1) { if (t < s) red[t] = fmaxf(red[t], red[t + s]); __syncthreads(); }
    float gm = red[0]; __syncthreads();
    float sum = 0.0f;
#pragma unroll
    for (int a = 0; a < 8; ++a) { v[a] = expf(v[a] - gm); sum += v[a]; }
    red[t] = sum; __syncthreads();
    for (int s = 128; s > 0; s >>= 1) { if (t < s) red[t] += red[t + s]; __syncthreads(); }
    float inv = 1.0f / red[0];
    float* o = out + (size_t)g * OUT_W;
#pragma unroll
    for (int a = 0; a < 8; ++a) o[2 + a * 256 + t] = v[a] * inv;
    if (t < 2) o[t] = 0.0f;
}

// ------------------------------------------------------------------- launcher
extern "C" void kernel_launch(void* const* d_in, const int* in_sizes, int n_in,
                              void* d_out, int out_size, void* d_ws, size_t ws_size,
                              hipStream_t stream) {
    const float* x   = (const float*)d_in[0];
    const int*   ei  = (const int*)d_in[1];
    const float* W1  = (const float*)d_in[3];
    const float* b1  = (const float*)d_in[4];
    const float* W2  = (const float*)d_in[5];
    const float* b2  = (const float*)d_in[6];
    const float* Wo1 = (const float*)d_in[7];
    const float* bo1 = (const float*)d_in[8];
    const float* Wo2 = (const float*)d_in[9];
    const float* bo2 = (const float*)d_in[10];
    const float* Wo3 = (const float*)d_in[11];
    const float* bo3 = (const float*)d_in[12];

    const int E = in_sizes[1] / 2;
    const int* src = ei;
    const int* dst = ei + E;

    char* ws = (char*)d_ws; size_t off = 0;
    auto take = [&](size_t bytes) -> char* {
        char* p = ws + off; off = (off + bytes + 255) & ~(size_t)255; return p;
    };
    float* dinv   = (float*)take((size_t)N_NODES * 4);
    float* bufC   = (float*)take((size_t)N_NODES * H1 * 4);   // 32 MB
    float* bufAgg = (float*)take((size_t)N_NODES * H1 * 4);   // 32 MB
    bf16*  P0     = (bf16*) take((size_t)N_NODES * H1 * 2);   // 16 MB ping
    bf16*  P1     = (bf16*) take((size_t)N_NODES * H1 * 2);   // 16 MB pong
    bf16*  W1b    = (bf16*) take((size_t)256 * 128 * 2);
    bf16*  W2b    = (bf16*) take((size_t)64  * 256 * 2);
    bf16*  Wo1b   = (bf16*) take((size_t)256 * 64  * 2);
    bf16*  Wo2b   = (bf16*) take((size_t)256 * 256 * 2);
    bf16*  Wo3b   = (bf16*) take((size_t)64  * 256 * 2);
    float* bo3p   = (float*)take(64 * 4);

    // --- conversions: x -> bf16, weights -> bf16 transposed (N-major), pad Wo3
    k_cast_bf16<<<(N_NODES * F_IN + 255) / 256, 256, 0, stream>>>(x, P0, N_NODES * F_IN);
    k_transpose_cast<<<(256 * 128 + 255) / 256, 256, 0, stream>>>(W1,  W1b, 128, 256, 256);
    k_transpose_cast<<<(64 * 256 + 255) / 256, 256, 0, stream>>>(W2,  W2b, 256, 64, 64);
    k_transpose_cast<<<(256 * 64 + 255) / 256, 256, 0, stream>>>(Wo1, Wo1b, 64, 256, 256);
    k_transpose_cast<<<(256 * 256 + 255) / 256, 256, 0, stream>>>(Wo2, Wo2b, 256, 256, 256);
    k_transpose_cast<<<(64 * 256 + 255) / 256, 256, 0, stream>>>(Wo3, Wo3b, 256, 8, 64);
    k_pad_bias<<<1, 256, 0, stream>>>(bo3, bo3p, 8, 64);

    // --- degrees: deg = 1 (self-loop) + in-degree; dinv = rsqrt(deg)
    k_fill<<<N_NODES / 256, 256, 0, stream>>>(dinv, 1.0f, N_NODES);
    k_deg_add<<<(E + 255) / 256, 256, 0, stream>>>(dst, E, dinv);
    k_rsqrt_inplace<<<N_NODES / 256, 256, 0, stream>>>(dinv, N_NODES);

    // --- GCN layer 1: g = (x@W1)*dinv[row]  (dual write: bufC = read copy, bufAgg = acc)
    k_gemm_bf16<4, true, false, false, true, true, false>
        <<<dim3(H1 / 128, N_NODES / 64), 128, 0, stream>>>(
            P0, W1b, N_NODES, H1, F_IN, dinv, nullptr, bufC, bufAgg, nullptr);
    k_edge_scatter<<<dim3(E / 256, H1 / 16), 256, 0, stream>>>(src, dst, bufC, bufAgg, H1);
    k_post1<<<N_NODES, H1, 0, stream>>>(bufAgg, dinv, b1, P1);

    // --- GCN layer 2
    k_gemm_bf16<2, true, false, false, true, true, false>
        <<<dim3(H2 / 64, N_NODES / 64), 128, 0, stream>>>(
            P1, W2b, N_NODES, H2, H1, dinv, nullptr, bufC, bufAgg, nullptr);
    k_edge_scatter<<<dim3(E / 256, H2 / 16), 256, 0, stream>>>(src, dst, bufC, bufAgg, H2);
    k_extract_hosts<<<(N_HOSTS * H2) / 256, 256, 0, stream>>>(bufAgg, dinv, b2, P0);

    // --- MLP head
    k_gemm_bf16<4, false, true, true, false, false, true>
        <<<dim3(H1 / 128, N_HOSTS / 64), 128, 0, stream>>>(
            P0, Wo1b, N_HOSTS, H1, H2, nullptr, bo1, nullptr, nullptr, P1);
    k_gemm_bf16<4, false, true, true, false, false, true>
        <<<dim3(H1 / 128, N_HOSTS / 64), 128, 0, stream>>>(
            P1, Wo2b, N_HOSTS, H1, H1, nullptr, bo2, nullptr, nullptr, P0);
    k_gemm_bf16<2, false, true, false, true, false, false>
        <<<dim3(64 / 64, N_HOSTS / 64), 128, 0, stream>>>(
            P0, Wo3b, N_HOSTS, 64, H1, nullptr, bo3p, bufC, nullptr, nullptr);

    // --- transpose-reshape + softmax + leading zeros
    k_softmax<<<N_GRAPHS, 256, 0, stream>>>(bufC, (float*)d_out);
}
